// Model_91259465106020
// MI455X (gfx1250) — matrix operand order repaired and checked by reading, not compiled, not run on hardware
//
#include <hip/hip_runtime.h>
#include <hip/hip_bf16.h>
#include <cstdint>
#include <cstddef>

#define B_    128
#define L_    128
#define FEAT_ 36
#define XTR_  12
#define EMB_  6
#define H_    1024
#define A_    256
#define TOUT_ 24
#define MROWS 152   // 151 attention rows, padded to 152
#define KCH   512   // K-chunk staged in LDS (64 rows x 512 halves = 64 KB)

typedef _Float16 half8  __attribute__((ext_vector_type(8)));
typedef _Float16 half16 __attribute__((ext_vector_type(16)));
typedef float    float8 __attribute__((ext_vector_type(8)));

// ---------------- WMMA tile loaders (CDNA5 16-bit layouts) ----------------
// A (16x32, MxK): lanes 0-15 = rows, K 0..15;
//                 lanes 16-31 = same rows, K 16..31 (contiguous 16 halves).
__device__ __forceinline__ half16 load_a16(const _Float16* base, int stride,
                                           int lane, int kk) {
  const _Float16* p = base + (size_t)(lane & 15) * stride + kk + ((lane & 16) ? 16 : 0);
  union { half16 v; half8 h[2]; } u;
  u.h[0] = *(const half8*)(p);
  u.h[1] = *(const half8*)(p + 8);
  return u.v;
}
// B (32x16, KxN) from row-major W[N][K]: lane%16 = column n,
// lanes 0-15 hold K 0..15, lanes 16-31 hold K 16..31 (contiguous 16 halves).
__device__ __forceinline__ half16 load_b16(const _Float16* base, int stride,
                                           int lane, int kk) {
  const _Float16* p = base + (size_t)(lane & 15) * stride + kk + ((lane & 16) ? 16 : 0);
  union { half16 v; half8 h[2]; } u;
  u.h[0] = *(const half8*)(p);
  u.h[1] = *(const half8*)(p + 8);
  return u.v;
}
// B tile read from the LDS-staged panel: row = g*16 + (lane&15), KCH-half rows.
__device__ __forceinline__ half16 lds_b16(const _Float16* sB, int lane, int g, int kl) {
  const _Float16* p = sB + (size_t)(g * 16 + (lane & 15)) * KCH + kl + ((lane & 16) ? 16 : 0);
  union { half16 v; half8 h[2]; } u;
  u.h[0] = *(const half8*)(p);
  u.h[1] = *(const half8*)(p + 8);
  return u.v;
}
__device__ __forceinline__ float8 wmma32(half16 a, half16 b, float8 c) {
  return __builtin_amdgcn_wmma_f32_16x16x32_f16(false, a, false, b, (short)0, c,
                                                false, false);
}
__device__ __forceinline__ float sigm(float x) { return 1.f / (1.f + expf(-x)); }

// Stage 64 weight rows x KCH halves into LDS with 128 threads:
// thread -> (row = tid>>1, 512B segment = tid&1).
__device__ __forceinline__ void stage_panel(_Float16* sB, const _Float16* W,
                                            int row_stride, int jt, int chunk) {
  int row = threadIdx.x >> 1, seg = threadIdx.x & 1;
  int g = row >> 4, rl = row & 15;
  const _Float16* src = W + (size_t)(g * H_ + jt * 16 + rl) * row_stride + chunk + seg * 256;
  _Float16* dst = sB + (size_t)row * KCH + seg * 256;
#pragma unroll
  for (int i = 0; i < 16; ++i)
    *(half8*)(dst + i * 16) = *(const half8*)(src + i * 16);
}

// ---------------- init / weight prep ----------------
__global__ void k_init(float* hF, float* cF, _Float16* h16) {
  int i = blockIdx.x * blockDim.x + threadIdx.x;
  if (i >= B_ * H_) return;
  hF[i] = 0.f; cF[i] = 0.f;
  h16[i] = (_Float16)0.f; h16[B_ * H_ + i] = (_Float16)0.f;
}

__global__ void k_prep(const float* eWih, const float* eWhh,
                       const float* dWih, const float* dWhh, const float* wk,
                       _Float16* Wih16, _Float16* Whh16, _Float16* decW16,
                       _Float16* Wk16) {
  int idx = blockIdx.x * blockDim.x + threadIdx.x;
  if (idx < 4096 * 3072) {
    int n = idx / 3072, k = idx - n * 3072;
    float v = (k < 2048) ? dWih[(size_t)n * 2048 + k] : dWhh[(size_t)n * 1024 + (k - 2048)];
    decW16[idx] = (_Float16)v;
  }
  if (idx < 4096 * 1024) Whh16[idx] = (_Float16)eWhh[idx];
  if (idx < 4096 * 32) {
    int n = idx >> 5, k = idx & 31;
    Wih16[idx] = (_Float16)((k < (EMB_ + XTR_)) ? eWih[n * (EMB_ + XTR_) + k] : 0.f);
  }
  if (idx < 256 * 1024) Wk16[idx] = (_Float16)wk[idx];
}

// ---------------- conv frontend ----------------
__global__ void k_conv1(const float* __restrict__ x, const float* __restrict__ w1,
                        const float* __restrict__ b1, float* __restrict__ h1) {
  int idx = blockIdx.x * blockDim.x + threadIdx.x;
  if (idx >= B_ * L_ * 34) return;
  int w = idx % 34, r = idx / 34, co = r & 127, b = r >> 7;
  float acc = b1[co];
  const float* xb = x + (size_t)b * L_ * (FEAT_ + XTR_);
  const float* wc = w1 + (size_t)co * L_ * 3;
  for (int ci = 0; ci < L_; ++ci) {
    const float* xr = xb + ci * (FEAT_ + XTR_) + w;
    const float* wr = wc + ci * 3;
    acc += xr[0] * wr[0] + xr[1] * wr[1] + xr[2] * wr[2];
  }
  h1[idx] = (acc > 20.f) ? acc : log1pf(expf(acc));   // softplus
}

__global__ void k_conv2(const float* __restrict__ h1, const float* __restrict__ w2,
                        const float* __restrict__ b2, float* __restrict__ h2) {
  int idx = blockIdx.x * blockDim.x + threadIdx.x;
  if (idx >= B_ * L_ * 32) return;
  int w = idx % 32, r = idx / 32, co = r & 127, b = r >> 7;
  float acc = b2[co];
  const float* hb = h1 + (size_t)b * L_ * 34;
  const float* wc = w2 + (size_t)co * L_ * 3;
  for (int ci = 0; ci < L_; ++ci) {
    const float* hr = hb + ci * 34 + w;
    const float* wr = wc + ci * 3;
    acc += hr[0] * wr[0] + hr[1] * wr[1] + hr[2] * wr[2];
  }
  h2[idx] = fmaxf(acc, 0.f);
}

// lin3 + concat extras -> lstm_in padded to K=32, f16
__global__ void k_lin3(const float* __restrict__ h2, const float* __restrict__ lw,
                       const float* __restrict__ lb, const float* __restrict__ x,
                       _Float16* __restrict__ li16) {
  int idx = blockIdx.x * blockDim.x + threadIdx.x;
  if (idx >= B_ * L_ * 32) return;
  int col = idx & 31, bl = idx >> 5;
  float v;
  if (col < EMB_) {
    const float* h2r = h2 + (size_t)bl * 32;
    const float* w = lw + col * 32;
    v = lb[col];
    for (int f = 0; f < 32; ++f) v += h2r[f] * w[f];
  } else if (col < EMB_ + XTR_) {
    v = x[(size_t)bl * (FEAT_ + XTR_) + FEAT_ + (col - EMB_)];
  } else {
    v = 0.f;
  }
  li16[idx] = (_Float16)v;
}

// ---------------- fused encoder LSTM step (WMMA + LDS-staged B) ----------------
// grid (64 hidden tiles, 2 batch halves), 128 threads (4 waves x 16 batch rows)
__global__ void k_enc_step(const _Float16* __restrict__ h16src,
                           _Float16* __restrict__ h16dst,
                           const _Float16* __restrict__ Whh16,
                           const _Float16* __restrict__ Wih16,
                           const _Float16* __restrict__ li16,
                           const float* __restrict__ bias,
                           float* __restrict__ hF, float* __restrict__ cF,
                           _Float16* __restrict__ buf16, int t) {
  __shared__ _Float16 sB[64 * KCH];           // 64 KB weight panel chunk
  const int lane = threadIdx.x & 31;
  const int mrow = blockIdx.y * 64 + ((threadIdx.x >> 5) << 4);  // batch row base
  const int jt   = blockIdx.x;                // hidden-column tile (0..63)

  float8 a0 = {0.f,0.f,0.f,0.f,0.f,0.f,0.f,0.f};
  float8 a1 = a0, a2 = a0, a3 = a0;
  const _Float16* Ah = h16src + (size_t)mrow * H_;

  for (int chunk = 0; chunk < H_; chunk += KCH) {
    __syncthreads();
    stage_panel(sB, Whh16, H_, jt, chunk);
    __syncthreads();
#pragma unroll
    for (int kl = 0; kl < KCH; kl += 32) {
      half16 av = load_a16(Ah, H_, lane, chunk + kl);
      a0 = wmma32(av, lds_b16(sB, lane, 0, kl), a0);
      a1 = wmma32(av, lds_b16(sB, lane, 1, kl), a1);
      a2 = wmma32(av, lds_b16(sB, lane, 2, kl), a2);
      a3 = wmma32(av, lds_b16(sB, lane, 3, kl), a3);
    }
  }

  // input (x_t @ Wih^T) term: K=18 padded to 32, one WMMA per gate
  half16 ax = load_a16(li16 + ((size_t)mrow * L_ + t) * 32, L_ * 32, lane, 0);
  a0 = wmma32(ax, load_b16(Wih16 + (size_t)(0 * H_ + jt * 16) * 32, 32, lane, 0), a0);
  a1 = wmma32(ax, load_b16(Wih16 + (size_t)(1 * H_ + jt * 16) * 32, 32, lane, 0), a1);
  a2 = wmma32(ax, load_b16(Wih16 + (size_t)(2 * H_ + jt * 16) * 32, 32, lane, 0), a2);
  a3 = wmma32(ax, load_b16(Wih16 + (size_t)(3 * H_ + jt * 16) * 32, 32, lane, 0), a3);

  const int n = jt * 16 + (lane & 15);
  const int bofs = mrow + ((lane & 16) ? 8 : 0);
#pragma unroll
  for (int j = 0; j < 8; ++j) {
    int b = bofs + j;
    float gi = a0[j] + bias[n];
    float gf = a1[j] + bias[H_ + n];
    float gg = a2[j] + bias[2 * H_ + n];
    float go = a3[j] + bias[3 * H_ + n];
    float cp = cF[(size_t)b * H_ + n];
    float cn = sigm(gf) * cp + sigm(gi) * tanhf(gg);
    float hn = sigm(go) * tanhf(cn);
    cF[(size_t)b * H_ + n] = cn;
    hF[(size_t)b * H_ + n] = hn;
    _Float16 hh = (_Float16)hn;
    h16dst[(size_t)b * H_ + n] = hh;
    buf16[((size_t)b * MROWS + t) * H_ + n] = hh;   // enc_out -> attention buffer
  }
}

// -------- fused decoder LSTM step (WMMA, K = 2048 ctx + 1024 h, LDS B) --------
__global__ void k_dec_step(const _Float16* __restrict__ ctx16,
                           const _Float16* __restrict__ h16src,
                           _Float16* __restrict__ h16dst,
                           const _Float16* __restrict__ W16,  // 4096 x 3072
                           const float* __restrict__ bias,
                           float* __restrict__ hF, float* __restrict__ cF,
                           _Float16* __restrict__ buf16, int t) {
  __shared__ _Float16 sB[64 * KCH];
  const int lane = threadIdx.x & 31;
  const int mrow = blockIdx.y * 64 + ((threadIdx.x >> 5) << 4);
  const int jt   = blockIdx.x;

  float8 a0 = {0.f,0.f,0.f,0.f,0.f,0.f,0.f,0.f};
  float8 a1 = a0, a2 = a0, a3 = a0;

  for (int chunk = 0; chunk < 3072; chunk += KCH) {
    __syncthreads();
    stage_panel(sB, W16, 3072, jt, chunk);
    __syncthreads();

    const _Float16* Ab;
    int astr, kbase;
    if (chunk < 2048) { Ab = ctx16 + (size_t)mrow * 3072; astr = 3072; kbase = chunk; }
    else              { Ab = h16src + (size_t)mrow * H_;  astr = H_;   kbase = chunk - 2048; }
#pragma unroll
    for (int kl = 0; kl < KCH; kl += 32) {
      half16 av = load_a16(Ab, astr, lane, kbase + kl);
      a0 = wmma32(av, lds_b16(sB, lane, 0, kl), a0);
      a1 = wmma32(av, lds_b16(sB, lane, 1, kl), a1);
      a2 = wmma32(av, lds_b16(sB, lane, 2, kl), a2);
      a3 = wmma32(av, lds_b16(sB, lane, 3, kl), a3);
    }
  }

  const int n = jt * 16 + (lane & 15);
  const int bofs = mrow + ((lane & 16) ? 8 : 0);
#pragma unroll
  for (int j = 0; j < 8; ++j) {
    int b = bofs + j;
    float gi = a0[j] + bias[n];
    float gf = a1[j] + bias[H_ + n];
    float gg = a2[j] + bias[2 * H_ + n];
    float go = a3[j] + bias[3 * H_ + n];
    float cp = cF[(size_t)b * H_ + n];
    float cn = sigm(gf) * cp + sigm(gi) * tanhf(gg);
    float hn = sigm(go) * tanhf(cn);
    cF[(size_t)b * H_ + n] = cn;
    hF[(size_t)b * H_ + n] = hn;
    _Float16 hh = (_Float16)hn;
    h16dst[(size_t)b * H_ + n] = hh;
    buf16[((size_t)b * MROWS + 128 + t) * H_ + n] = hh;  // append s for next steps
  }
}

// ---------------- attention pieces ----------------
// kp[b][m][a] = buf16[b][m][:] . Wk[a][:]  for m = 0..127 (WMMA, one-shot)
__global__ void k_kp_pre(const _Float16* __restrict__ buf16,
                         const _Float16* __restrict__ Wk16,
                         float* __restrict__ kp) {
  const int lane = threadIdx.x & 31;
  const int wave = threadIdx.x >> 5;   // m tile (0..7)
  const int at = blockIdx.x;           // a tile (0..15)
  const int b  = blockIdx.y;           // batch
  const _Float16* Ab = buf16 + ((size_t)b * MROWS + wave * 16) * H_;
  const _Float16* Bb = Wk16 + (size_t)(at * 16) * H_;
  float8 acc = {0.f,0.f,0.f,0.f,0.f,0.f,0.f,0.f};
  for (int kk = 0; kk < H_; kk += 32)
    acc = wmma32(load_a16(Ab, H_, lane, kk), load_b16(Bb, H_, lane, kk), acc);
  const int a = at * 16 + (lane & 15);
  const int mb = wave * 16 + ((lane & 16) ? 8 : 0);
#pragma unroll
  for (int j = 0; j < 8; ++j)
    kp[((size_t)b * MROWS + mb + j) * A_ + a] = acc[j];
}

// append kp row 128+t from new decoder h (WMMA over batch rows)
__global__ void k_kp_app(const _Float16* __restrict__ h16new,
                         const _Float16* __restrict__ Wk16,
                         float* __restrict__ kp, int t) {
  const int lane = threadIdx.x & 31;
  const int wave = threadIdx.x >> 5;   // batch tile (0..7)
  const int at = blockIdx.x;           // a tile (0..15)
  const _Float16* Ab = h16new + (size_t)(wave * 16) * H_;
  const _Float16* Bb = Wk16 + (size_t)(at * 16) * H_;
  float8 acc = {0.f,0.f,0.f,0.f,0.f,0.f,0.f,0.f};
  for (int kk = 0; kk < H_; kk += 32)
    acc = wmma32(load_a16(Ab, H_, lane, kk), load_b16(Bb, H_, lane, kk), acc);
  const int a = at * 16 + (lane & 15);
  const int bb = wave * 16 + ((lane & 16) ? 8 : 0);
#pragma unroll
  for (int j = 0; j < 8; ++j)
    kp[((size_t)(bb + j) * MROWS + 128 + t) * A_ + a] = acc[j];
}

// qp[b][n][a] = q[b][n][:] . Wq[a][:], q = stack(h, c)
__global__ void k_qp(const float* __restrict__ hF, const float* __restrict__ cF,
                     const float* __restrict__ Wq, float* __restrict__ qp) {
  int idx = blockIdx.x * blockDim.x + threadIdx.x;
  if (idx >= B_ * 2 * A_) return;
  int a = idx % A_, bn = idx / A_, nq = bn & 1, b = bn >> 1;
  const float* src = (nq == 0 ? hF : cF) + (size_t)b * H_;
  const float* w = Wq + (size_t)a * H_;
  float acc = 0.f;
  for (int k = 0; k < H_; ++k) acc += src[k] * w[k];
  qp[(size_t)(b * 2 + nq) * A_ + a] = acc;
}

// scores + softmax over the 2 query entries (axis=1 in reference)
__global__ void k_scores(const float* __restrict__ qp, const float* __restrict__ kp,
                         const float* __restrict__ wv, float* __restrict__ wgt,
                         int Mt) {
  int idx = blockIdx.x * blockDim.x + threadIdx.x;
  if (idx >= B_ * Mt) return;
  int b = idx / Mt, m = idx % Mt;
  const float* q0 = qp + (size_t)(b * 2 + 0) * A_;
  const float* q1 = qp + (size_t)(b * 2 + 1) * A_;
  const float* kr = kp + ((size_t)b * MROWS + m) * A_;
  float s0 = 0.f, s1 = 0.f;
  for (int a = 0; a < A_; ++a) {
    float kv = kr[a], w = wv[a];
    s0 += w * tanhf(q0[a] + kv);
    s1 += w * tanhf(q1[a] + kv);
  }
  float mx = fmaxf(s0, s1);
  float e0 = expf(s0 - mx), e1 = expf(s1 - mx);
  float inv = 1.f / (e0 + e1);
  wgt[(size_t)(b * 2 + 0) * MROWS + m] = e0 * inv;
  wgt[(size_t)(b * 2 + 1) * MROWS + m] = e1 * inv;
}

// ctx[b][n][hid] = sum_m wgt * buf  -> f16 into combined decoder A matrix
__global__ void k_ctx(const float* __restrict__ wgt, const _Float16* __restrict__ buf16,
                      _Float16* __restrict__ ctx16, int Mt) {
  int idx = blockIdx.x * blockDim.x + threadIdx.x;
  if (idx >= B_ * 2 * H_) return;
  int hid = idx & (H_ - 1), bn = idx >> 10, nq = bn & 1, b = bn >> 1;
  const float* w = wgt + (size_t)(b * 2 + nq) * MROWS;
  const _Float16* bp = buf16 + (size_t)b * MROWS * H_ + hid;
  float acc = 0.f;
  for (int m = 0; m < Mt; ++m) acc += w[m] * (float)bp[(size_t)m * H_];
  ctx16[(size_t)b * 3072 + nq * H_ + hid] = (_Float16)acc;
}

// one wave32 per batch row, shuffle reduction
__global__ void k_out(const float* __restrict__ hF, const float* __restrict__ ow,
                      const float* __restrict__ ob, float* __restrict__ out, int t) {
  int lane = threadIdx.x & 31;
  int b = (blockIdx.x << 3) + (threadIdx.x >> 5);
  if (b >= B_) return;
  const float* h = hF + (size_t)b * H_;
  float acc = 0.f;
  for (int k = lane; k < H_; k += 32) acc += h[k] * ow[k];
#pragma unroll
  for (int s = 16; s > 0; s >>= 1) acc += __shfl_xor(acc, s, 32);
  if (lane == 0) out[b * TOUT_ + t] = acc + ob[0];
}

// ---------------- host ----------------
extern "C" void kernel_launch(void* const* d_in, const int* in_sizes, int n_in,
                              void* d_out, int out_size, void* d_ws, size_t ws_size,
                              hipStream_t stream) {
  const float* x       = (const float*)d_in[0];
  const float* conv1_w = (const float*)d_in[1];
  const float* conv1_b = (const float*)d_in[2];
  const float* conv2_w = (const float*)d_in[3];
  const float* conv2_b = (const float*)d_in[4];
  const float* lin3_w  = (const float*)d_in[5];
  const float* lin3_b  = (const float*)d_in[6];
  const float* enc_Wih = (const float*)d_in[7];
  const float* enc_Whh = (const float*)d_in[8];
  const float* enc_b   = (const float*)d_in[9];
  const float* attn_Wq = (const float*)d_in[10];
  const float* attn_Wk = (const float*)d_in[11];
  const float* attn_Wv = (const float*)d_in[12];
  const float* dec_Wih = (const float*)d_in[13];
  const float* dec_Whh = (const float*)d_in[14];
  const float* dec_b   = (const float*)d_in[15];
  const float* out_w   = (const float*)d_in[16];
  const float* out_b   = (const float*)d_in[17];
  float* out = (float*)d_out;

  char* ws = (char*)d_ws;
  size_t off = 0;
  auto alloc = [&](size_t n) { size_t o = off; off = (off + n + 255) & ~(size_t)255; return o; };

  float*     h1     = (float*)    (ws + alloc((size_t)B_ * L_ * 34 * 4));
  float*     h2     = (float*)    (ws + alloc((size_t)B_ * L_ * 32 * 4));
  _Float16*  li16   = (_Float16*) (ws + alloc((size_t)B_ * L_ * 32 * 2));
  _Float16*  Whh16  = (_Float16*) (ws + alloc((size_t)4096 * 1024 * 2));
  _Float16*  Wih16  = (_Float16*) (ws + alloc((size_t)4096 * 32 * 2));
  _Float16*  decW16 = (_Float16*) (ws + alloc((size_t)4096 * 3072 * 2));
  _Float16*  Wk16   = (_Float16*) (ws + alloc((size_t)256 * 1024 * 2));
  _Float16*  buf16  = (_Float16*) (ws + alloc((size_t)B_ * MROWS * H_ * 2));
  float*     hF     = (float*)    (ws + alloc((size_t)B_ * H_ * 4));
  float*     cF     = (float*)    (ws + alloc((size_t)B_ * H_ * 4));
  _Float16*  h16    = (_Float16*) (ws + alloc((size_t)2 * B_ * H_ * 2));   // ping-pong
  float*     kp     = (float*)    (ws + alloc((size_t)B_ * MROWS * A_ * 4));
  float*     qp     = (float*)    (ws + alloc((size_t)B_ * 2 * A_ * 4));
  float*     wgt    = (float*)    (ws + alloc((size_t)B_ * 2 * MROWS * 4));
  _Float16*  ctx16  = (_Float16*) (ws + alloc((size_t)B_ * 3072 * 2));

  const int TPB = 256;
  // init state + convert weights to f16 (decW is cat[Wih|Whh], Wih padded 18->32)
  k_init<<<(B_ * H_ + TPB - 1) / TPB, TPB, 0, stream>>>(hF, cF, h16);
  k_prep<<<(4096 * 3072 + TPB - 1) / TPB, TPB, 0, stream>>>(
      enc_Wih, enc_Whh, dec_Wih, dec_Whh, attn_Wk, Wih16, Whh16, decW16, Wk16);

  // conv frontend
  k_conv1<<<(B_ * L_ * 34 + TPB - 1) / TPB, TPB, 0, stream>>>(x, conv1_w, conv1_b, h1);
  k_conv2<<<(B_ * L_ * 32 + TPB - 1) / TPB, TPB, 0, stream>>>(h1, conv2_w, conv2_b, h2);
  k_lin3<<<(B_ * L_ * 32 + TPB - 1) / TPB, TPB, 0, stream>>>(h2, lin3_w, lin3_b, x, li16);

  // encoder: 128 fused WMMA LSTM steps (ping-pong f16 h)
  int cur = 0;
  for (int t = 0; t < L_; ++t) {
    k_enc_step<<<dim3(64, 2), 128, 0, stream>>>(h16 + (size_t)cur * B_ * H_,
                                                h16 + (size_t)(cur ^ 1) * B_ * H_,
                                                Whh16, Wih16, li16, enc_b, hF, cF,
                                                buf16, t);
    cur ^= 1;
  }

  // kp for the 128 encoder rows (rows 0..126 static + row 127 = s0)
  k_kp_pre<<<dim3(16, B_), 256, 0, stream>>>(buf16, Wk16, kp);

  // decoder: 24 steps
  for (int t = 0; t < TOUT_; ++t) {
    int Mt = L_ + t;  // valid attention rows 0..127+t
    k_qp<<<(B_ * 2 * A_ + TPB - 1) / TPB, TPB, 0, stream>>>(hF, cF, attn_Wq, qp);
    k_scores<<<(B_ * Mt + TPB - 1) / TPB, TPB, 0, stream>>>(qp, kp, attn_Wv, wgt, Mt);
    k_ctx<<<(B_ * 2 * H_ + TPB - 1) / TPB, TPB, 0, stream>>>(wgt, buf16, ctx16, Mt);
    k_dec_step<<<dim3(64, 2), 128, 0, stream>>>(ctx16,
                                                h16 + (size_t)cur * B_ * H_,
                                                h16 + (size_t)(cur ^ 1) * B_ * H_,
                                                decW16, dec_b, hF, cF, buf16, t);
    cur ^= 1;
    k_out<<<16, 256, 0, stream>>>(hF, out_w, out_b, out, t);
    k_kp_app<<<16, 256, 0, stream>>>(h16 + (size_t)cur * B_ * H_, Wk16, kp, t);
  }
}
